// MLLABlock_38233798869630
// MI455X (gfx1250) — matrix operand order repaired and checked by reading, not compiled, not run on hardware
//
#include <hip/hip_runtime.h>
#include <cstdint>
#include <cstddef>

// Problem constants (B,C,H,W) = (16,256,56,56), heads=8
#define BN   16
#define CC   256
#define HH   56
#define WWD  56
#define SPX  (HH*WWD)      // 3136
#define HS   28
#define WS2  28
#define SPS  (HS*WS2)      // 784
#define NTOK (BN*SPX)      // 50176
#define NTOKS (BN*SPS)     // 12544
#define NH   8
#define HD   32

typedef __attribute__((ext_vector_type(16))) __bf16   v16bf;
typedef __attribute__((ext_vector_type(8)))  float    v8f;
typedef __attribute__((ext_vector_type(4)))  unsigned int u32x4;

__device__ __forceinline__ unsigned short f2bf(float f) {
    unsigned int u = __float_as_uint(f);
    u += 0x7FFFu + ((u >> 16) & 1u);   // round-to-nearest-even
    return (unsigned short)(u >> 16);
}

// ---------------- WMMA fragment loaders (bf16, 16x16x32) ----------------
// A 16x32: lane l -> row (l&15); halves 0..7 = K at (l>>4)*8 .. +7, halves 8..15 = +16
__device__ __forceinline__ v16bf load_fragA(const unsigned short* base, int ld, int lane) {
    const int r  = lane & 15;
    const int ko = (lane >> 4) * 8;
    const unsigned short* p = base + (size_t)r * ld + ko;
    union { u32x4 u[2]; v16bf v; } t;
    t.u[0] = *(const u32x4*)(p);
    t.u[1] = *(const u32x4*)(p + 16);
    return t.v;
}
// B 32x16 (from K-major weights): lane l -> col N=(l&15); halves 0..7 = K at (l>>4)*8 .. +7, halves 8..15 = +16
__device__ __forceinline__ v16bf load_fragB(const unsigned short* base, int ld, int lane) {
    const unsigned short* p = base + (lane & 15) + (size_t)((lane >> 4) * 8) * ld;
    union { unsigned short s[16]; v16bf v; } t;
#pragma unroll
    for (int i = 0; i < 8; ++i) {
        t.s[i]     = p[(size_t)i * ld];
        t.s[i + 8] = p[(size_t)(i + 16) * ld];
    }
    return t.v;
}

// ---------------- generic depthwise conv (+optional residual) ----------------
__global__ void dwconv_kernel(const float* __restrict__ in, const float* __restrict__ w,
                              const float* __restrict__ bias, const float* __restrict__ resid,
                              float* __restrict__ out,
                              int B, int C, int Hi, int Wi, int Ho, int Wo,
                              int ks, int stride, int pad) {
    size_t idx = (size_t)blockIdx.x * blockDim.x + threadIdx.x;
    size_t total = (size_t)B * C * Ho * Wo;
    if (idx >= total) return;
    int wo = (int)(idx % Wo);
    int ho = (int)((idx / Wo) % Ho);
    int c  = (int)((idx / ((size_t)Wo * Ho)) % C);
    int b  = (int)(idx / ((size_t)Wo * Ho * C));
    float acc = bias[c];
    const float* ip = in + ((size_t)b * C + c) * Hi * Wi;
    const float* wp = w + (size_t)c * ks * ks;
    for (int ky = 0; ky < ks; ++ky) {
        int yi = ho * stride - pad + ky;
        if (yi < 0 || yi >= Hi) continue;
        for (int kx = 0; kx < ks; ++kx) {
            int xi = wo * stride - pad + kx;
            if (xi < 0 || xi >= Wi) continue;
            acc += ip[(size_t)yi * Wi + xi] * wp[ky * ks + kx];
        }
    }
    if (resid) acc += resid[idx];
    out[idx] = acc;
}

// agg_conv: 5x5 dw conv over (c_ + nearest-up2(attn)) fused
__global__ void dwconv_up_kernel(const float* __restrict__ cb, const float* __restrict__ dsm,
                                 const float* __restrict__ w, const float* __restrict__ bias,
                                 float* __restrict__ out) {
    size_t idx = (size_t)blockIdx.x * blockDim.x + threadIdx.x;
    size_t total = (size_t)BN * CC * SPX;
    if (idx >= total) return;
    int x = (int)(idx % WWD);
    int y = (int)((idx / WWD) % HH);
    int c = (int)((idx / SPX) % CC);
    int b = (int)(idx / ((size_t)SPX * CC));
    const float* cp = cb  + ((size_t)b * CC + c) * SPX;
    const float* dp = dsm + ((size_t)b * CC + c) * SPS;
    const float* wp = w + (size_t)c * 25;
    float acc = bias[c];
    for (int ky = 0; ky < 5; ++ky) {
        int yi = y - 2 + ky;
        if (yi < 0 || yi >= HH) continue;
        for (int kx = 0; kx < 5; ++kx) {
            int xi = x - 2 + kx;
            if (xi < 0 || xi >= WWD) continue;
            float v = cp[yi * WWD + xi] + dp[(yi >> 1) * WS2 + (xi >> 1)];
            acc += v * wp[ky * 5 + kx];
        }
    }
    out[idx] = acc;
}

// ---------------- layernorm over C (NCHW in) -> token-major bf16 ----------------
__global__ void layernorm_pack_kernel(const float* __restrict__ in, const float* __restrict__ gam,
                                      const float* __restrict__ bet, unsigned short* __restrict__ out,
                                      int SP) {
    int wid  = threadIdx.x >> 5;
    int lane = threadIdx.x & 31;
    int tok  = blockIdx.x * (blockDim.x >> 5) + wid;
    if (tok >= BN * SP) return;
    int b = tok / SP, sp = tok - b * SP;
    const float* base = in + ((size_t)b * CC) * SP + sp;
    float v[8];
    float s = 0.f, s2 = 0.f;
#pragma unroll
    for (int i = 0; i < 8; ++i) {
        float x = base[(size_t)(lane + 32 * i) * SP];
        v[i] = x; s += x; s2 += x * x;
    }
#pragma unroll
    for (int m = 16; m >= 1; m >>= 1) {
        s  += __shfl_xor(s,  m, 32);
        s2 += __shfl_xor(s2, m, 32);
    }
    float mu  = s * (1.0f / CC);
    float var = s2 * (1.0f / CC) - mu * mu;
    float rs  = rsqrtf(var + 1e-5f);
    unsigned short* o = out + (size_t)tok * CC;
#pragma unroll
    for (int i = 0; i < 8; ++i) {
        int c = lane + 32 * i;
        o[c] = f2bf((v[i] - mu) * rs * gam[c] + bet[c]);
    }
}

// NCHW f32 -> token-major bf16
__global__ void pack_tok_kernel(const float* __restrict__ in, unsigned short* __restrict__ out, int SP) {
    size_t idx = (size_t)blockIdx.x * blockDim.x + threadIdx.x;
    size_t total = (size_t)BN * CC * SP;
    if (idx >= total) return;
    int sp = (int)(idx % SP);
    int c  = (int)((idx / SP) % CC);
    int b  = (int)(idx / ((size_t)SP * CC));
    out[((size_t)b * SP + sp) * CC + c] = f2bf(in[idx]);
}

// silu(g)*a (NCHW) -> token-major bf16
__global__ void pack_silu_mul_kernel(const float* __restrict__ g, const float* __restrict__ a,
                                     unsigned short* __restrict__ out) {
    size_t idx = (size_t)blockIdx.x * blockDim.x + threadIdx.x;
    size_t total = (size_t)BN * CC * SPX;
    if (idx >= total) return;
    int sp = (int)(idx % SPX);
    int c  = (int)((idx / SPX) % CC);
    int b  = (int)(idx / ((size_t)SPX * CC));
    float gv = g[idx];
    float sv = gv / (1.f + __expf(-gv));
    out[((size_t)b * SPX + sp) * CC + c] = f2bf(sv * a[idx]);
}

// f32 [N][K] weights -> bf16 K-major [K][N]
__global__ void convert_w_kernel(const float* __restrict__ w, unsigned short* __restrict__ out,
                                 int N, int K) {
    size_t i = (size_t)blockIdx.x * blockDim.x + threadIdx.x;
    if (i >= (size_t)N * K) return;
    int n = (int)(i / K), k = (int)(i % K);
    out[(size_t)k * N + n] = f2bf(w[i]);
}

// ---------------- linear attention (per (b,head): d=e=32, n=784) ----------------
__global__ void attn_kv_kernel(const float* __restrict__ qkb, const float* __restrict__ dsm,
                               float* __restrict__ kv, float* __restrict__ km) {
    int bh = blockIdx.x;
    int b = bh >> 3, h = bh & 7;
    const float* kb = qkb + ((size_t)b * 512 + 256 + h * HD) * SPS;  // k = split 1
    const float* vb = dsm + ((size_t)b * CC + h * HD) * SPS;
    int t  = threadIdx.x;           // 256 threads
    int d  = t >> 3;                // 0..31
    int e0 = (t & 7) * 4;           // 0,4,...,28
    const float* kd = kb + (size_t)d * SPS;
    const float* v0 = vb + (size_t)(e0 + 0) * SPS;
    const float* v1 = vb + (size_t)(e0 + 1) * SPS;
    const float* v2 = vb + (size_t)(e0 + 2) * SPS;
    const float* v3 = vb + (size_t)(e0 + 3) * SPS;
    float s0 = 0.f, s1 = 0.f, s2 = 0.f, s3 = 0.f;
    for (int n = 0; n < SPS; ++n) {
        float kk = kd[n];
        s0 += kk * v0[n]; s1 += kk * v1[n]; s2 += kk * v2[n]; s3 += kk * v3[n];
    }
    const float inv = 1.0f / (float)SPS;   // s^2 = 1/n from k*s, v*s
    float* kvb = kv + (size_t)bh * HD * HD;
    kvb[d * HD + e0 + 0] = s0 * inv;
    kvb[d * HD + e0 + 1] = s1 * inv;
    kvb[d * HD + e0 + 2] = s2 * inv;
    kvb[d * HD + e0 + 3] = s3 * inv;
    if (t < HD) {
        const float* kr = kb + (size_t)t * SPS;
        float m = 0.f;
        for (int n = 0; n < SPS; ++n) m += kr[n];
        km[(size_t)bh * HD + t] = m * inv;  // k.mean over n
    }
}

__global__ void attn_out_kernel(const float* __restrict__ qkb, const float* __restrict__ kv,
                                const float* __restrict__ km, float* __restrict__ outb) {
    __shared__ float skv[HD * HD];
    __shared__ float skm[HD];
    int bh = blockIdx.x;
    int b = bh >> 3, h = bh & 7;
    for (int i = threadIdx.x; i < HD * HD; i += blockDim.x) skv[i] = kv[(size_t)bh * HD * HD + i];
    if (threadIdx.x < HD) skm[threadIdx.x] = km[(size_t)bh * HD + threadIdx.x];
    __syncthreads();
    const float* qb = qkb + ((size_t)b * 512 + h * HD) * SPS;   // q = split 0
    float* ob = outb + ((size_t)b * CC + h * HD) * SPS;
    for (int n = threadIdx.x; n < SPS; n += blockDim.x) {
        float q[HD];
        float den = 1e-6f;
#pragma unroll
        for (int d = 0; d < HD; ++d) { q[d] = qb[(size_t)d * SPS + n]; den += q[d] * skm[d]; }
        float rd = 1.0f / den;
#pragma unroll 4
        for (int e = 0; e < HD; ++e) {
            float s = 0.f;
#pragma unroll
            for (int d = 0; d < HD; ++d) s += q[d] * skv[d * HD + e];
            ob[(size_t)e * SPS + n] = s * rd;
        }
    }
}

// ---------------- WMMA GEMM: out[M,N] = A[M,K] * Wt[K,N] + bias, fused epilogues --------
// mode 0: iproj  -> split channels to out0/out1 (NCHW f32)
// mode 1: qk     -> elu(x)+1, store channel-major [b, N, SP]
// mode 2: +res   -> add NCHW residual, store NCHW f32
// mode 3: gelu   -> exact gelu, store bf16 token-major [M, N]
__global__ __launch_bounds__(256) void gemm_wmma_kernel(
        const unsigned short* __restrict__ A, const unsigned short* __restrict__ Wt,
        const float* __restrict__ bias,
        int M, int N, int lda, int k_per_group, int n_per_group,
        int mode, int SP,
        float* __restrict__ out0, float* __restrict__ out1,
        const float* __restrict__ res, unsigned short* __restrict__ outh) {
    const int lane = threadIdx.x & 31;
    const int wave = threadIdx.x >> 5;
    const int wm = wave >> 1, wn = wave & 1;
    const int m_base = blockIdx.y * 128 + wm * 32;
    const int n_base = blockIdx.x * 64 + wn * 32;
    const int grp = n_base / n_per_group;
    const size_t acol = (size_t)grp * k_per_group;

    v8f acc00 = {0.f,0.f,0.f,0.f,0.f,0.f,0.f,0.f};
    v8f acc01 = acc00, acc10 = acc00, acc11 = acc00;

    const unsigned short* Ab0 = A + (size_t)m_base * lda + acol;
    const unsigned short* Ab1 = Ab0 + (size_t)16 * lda;

    for (int k0 = 0; k0 < k_per_group; k0 += 32) {
        v16bf a0 = load_fragA(Ab0 + k0, lda, lane);
        v16bf a1 = load_fragA(Ab1 + k0, lda, lane);
        const unsigned short* Bb = Wt + (size_t)k0 * N + n_base;
        v16bf b0 = load_fragB(Bb,      N, lane);
        v16bf b1 = load_fragB(Bb + 16, N, lane);
        acc00 = __builtin_amdgcn_wmma_f32_16x16x32_bf16(false, a0, false, b0, (short)0, acc00, false, false);
        acc01 = __builtin_amdgcn_wmma_f32_16x16x32_bf16(false, a0, false, b1, (short)0, acc01, false, false);
        acc10 = __builtin_amdgcn_wmma_f32_16x16x32_bf16(false, a1, false, b0, (short)0, acc10, false, false);
        acc11 = __builtin_amdgcn_wmma_f32_16x16x32_bf16(false, a1, false, b1, (short)0, acc11, false, false);
    }

    const int col  = lane & 15;
    const int radd = (lane >> 4) * 8;
    v8f accs[4] = {acc00, acc01, acc10, acc11};
#pragma unroll
    for (int t = 0; t < 4; ++t) {
        int mi = t >> 1, ni = t & 1;
        int gn = n_base + ni * 16 + col;
        float bv = bias[gn];
#pragma unroll
        for (int r = 0; r < 8; ++r) {
            int gm = m_base + mi * 16 + radd + r;
            float val = accs[t][r] + bv;
            if (mode == 0) {
                int b = gm / SP, sp = gm - b * SP;
                if (gn < CC) out0[((size_t)b * CC + gn) * SP + sp] = val;
                else         out1[((size_t)b * CC + (gn - CC)) * SP + sp] = val;
            } else if (mode == 1) {
                float e = (val > 0.f) ? (val + 1.f) : __expf(val);   // elu+1
                int b = gm / SP, sp = gm - b * SP;
                out0[((size_t)b * N + gn) * SP + sp] = e;
            } else if (mode == 2) {
                int b = gm / SP, sp = gm - b * SP;
                size_t o = ((size_t)b * N + gn) * SP + sp;
                out0[o] = val + res[o];
            } else {
                float gl = 0.5f * val * (1.f + erff(val * 0.70710678118f));
                outh[(size_t)gm * N + gn] = f2bf(gl);
            }
        }
    }
}

// ---------------------------------------------------------------------------
extern "C" void kernel_launch(void* const* d_in, const int* in_sizes, int n_in,
                              void* d_out, int out_size, void* d_ws, size_t ws_size,
                              hipStream_t stream) {
    (void)in_sizes; (void)n_in; (void)out_size; (void)ws_size;
    const float* x       = (const float*)d_in[0];
    const float* cpe1_w  = (const float*)d_in[1];
    const float* cpe1_b  = (const float*)d_in[2];
    const float* norm1_g = (const float*)d_in[3];
    const float* norm1_b = (const float*)d_in[4];
    const float* iproj_w = (const float*)d_in[5];
    const float* iproj_b = (const float*)d_in[6];
    const float* oproj_w = (const float*)d_in[7];
    const float* oproj_b = (const float*)d_in[8];
    const float* aggd_w  = (const float*)d_in[9];
    const float* aggd_b  = (const float*)d_in[10];
    const float* qk_w    = (const float*)d_in[11];
    const float* qk_b    = (const float*)d_in[12];
    const float* pe_w    = (const float*)d_in[13];
    const float* pe_b    = (const float*)d_in[14];
    const float* aggc_w  = (const float*)d_in[15];
    const float* aggc_b  = (const float*)d_in[16];
    const float* cpe2_w  = (const float*)d_in[17];
    const float* cpe2_b  = (const float*)d_in[18];
    const float* norm2_g = (const float*)d_in[19];
    const float* norm2_b = (const float*)d_in[20];
    const float* fc1_w   = (const float*)d_in[21];
    const float* fc1_b   = (const float*)d_in[22];
    const float* fc2_w   = (const float*)d_in[23];
    const float* fc2_b   = (const float*)d_in[24];
    float* out = (float*)d_out;

    char* ws = (char*)d_ws;
    size_t off = 0;
    auto alloc = [&](size_t bytes) -> char* {
        char* p = ws + off;
        off += (bytes + 255) & ~(size_t)255;
        return p;
    };
    const size_t FB  = (size_t)BN * CC * SPX * 4;   // full NCHW f32
    const size_t FBS = (size_t)BN * CC * SPS * 4;   // small NCHW f32
    float* x1   = (float*)alloc(FB);
    float* gb   = (float*)alloc(FB);
    float* cb   = (float*)alloc(FB);
    float* ab   = (float*)alloc(FB);
    float* x2   = (float*)alloc(FB);
    float* x3   = (float*)alloc(FB);
    float* dsm  = (float*)alloc(FBS);
    float* attn = (float*)alloc(FBS);
    float* qkb  = (float*)alloc((size_t)BN * 512 * SPS * 4);
    float* kv   = (float*)alloc((size_t)BN * NH * HD * HD * 4);
    float* km   = (float*)alloc((size_t)BN * NH * HD * 4);
    unsigned short* tok   = (unsigned short*)alloc((size_t)NTOK * CC * 2);
    unsigned short* dstok = (unsigned short*)alloc((size_t)NTOKS * CC * 2);
    unsigned short* hbuf  = (unsigned short*)alloc((size_t)NTOK * 1024 * 2);
    unsigned short* wip = (unsigned short*)alloc((size_t)512 * 256 * 2);
    unsigned short* wop = (unsigned short*)alloc((size_t)256 * 256 * 2);
    unsigned short* wqk = (unsigned short*)alloc((size_t)512 * 128 * 2);
    unsigned short* wf1 = (unsigned short*)alloc((size_t)1024 * 256 * 2);
    unsigned short* wf2 = (unsigned short*)alloc((size_t)256 * 1024 * 2);

    // one-time (per launch) weight conversions: f32 [N][K] -> bf16 K-major [K][N]
    convert_w_kernel<<<(512 * 256 + 255) / 256, 256, 0, stream>>>(iproj_w, wip, 512, 256);
    convert_w_kernel<<<(256 * 256 + 255) / 256, 256, 0, stream>>>(oproj_w, wop, 256, 256);
    convert_w_kernel<<<(512 * 128 + 255) / 256, 256, 0, stream>>>(qk_w,    wqk, 512, 128);
    convert_w_kernel<<<(1024 * 256 + 255) / 256, 256, 0, stream>>>(fc1_w,  wf1, 1024, 256);
    convert_w_kernel<<<(256 * 1024 + 255) / 256, 256, 0, stream>>>(fc2_w,  wf2, 256, 1024);

    const size_t totF = (size_t)BN * CC * SPX;
    const size_t totS = (size_t)BN * CC * SPS;
    const int gF = (int)((totF + 255) / 256);
    const int gS = (int)((totS + 255) / 256);

    // 1) x1 = x + dw5(x, cpe1)
    dwconv_kernel<<<gF, 256, 0, stream>>>(x, cpe1_w, cpe1_b, x, x1, BN, CC, HH, WWD, HH, WWD, 5, 1, 2);
    // 2) t = LN(x1) -> bf16 token-major
    layernorm_pack_kernel<<<NTOK / 8, 256, 0, stream>>>(x1, norm1_g, norm1_b, tok, SPX);
    // 3) gc = t @ iproj^T + b ; split -> g, c_
    gemm_wmma_kernel<<<dim3(512 / 64, NTOK / 128), 256, 0, stream>>>(
        tok, wip, iproj_b, NTOK, 512, 256, 256, 512, 0, SPX, gb, cb, nullptr, nullptr);
    // 4) dsm = dw5 stride2 (agg_down) of c_
    dwconv_kernel<<<gS, 256, 0, stream>>>(cb, aggd_w, aggd_b, nullptr, dsm, BN, CC, HH, WWD, HS, WS2, 5, 2, 2);
    // 5) token-major bf16 of dsm
    pack_tok_kernel<<<gS, 256, 0, stream>>>(dsm, dstok, SPS);
    // 6) qk = elu(grouped 1x1 conv)+1  (groups=2), channel-major store
    gemm_wmma_kernel<<<dim3(512 / 64, NTOKS / 128), 256, 0, stream>>>(
        dstok, wqk, qk_b, NTOKS, 512, 256, 128, 256, 1, SPS, qkb, nullptr, nullptr, nullptr);
    // 7) kv = (k*s)(v*s)^T, kmean
    attn_kv_kernel<<<BN * NH, 256, 0, stream>>>(qkb, dsm, kv, km);
    // 8) attn = (q^T kv) / (q . kmean + 1e-6)
    attn_out_kernel<<<BN * NH, 256, 0, stream>>>(qkb, kv, km, attn);
    // 9) attn += dw3(dsm, pe)
    dwconv_kernel<<<gS, 256, 0, stream>>>(dsm, pe_w, pe_b, attn, attn, BN, CC, HS, WS2, HS, WS2, 3, 1, 1);
    // 10) a = dw5(c_ + up2(attn), agg_conv)
    dwconv_up_kernel<<<gF, 256, 0, stream>>>(cb, attn, aggc_w, aggc_b, ab);
    // 11) s = silu(g)*a -> bf16 token-major
    pack_silu_mul_kernel<<<gF, 256, 0, stream>>>(gb, ab, tok);
    // 12) x2 = x1 + oproj(s)
    gemm_wmma_kernel<<<dim3(256 / 64, NTOK / 128), 256, 0, stream>>>(
        tok, wop, oproj_b, NTOK, 256, 256, 256, 256, 2, SPX, x2, nullptr, x1, nullptr);
    // 13) x3 = x2 + dw3(x2, cpe2)
    dwconv_kernel<<<gF, 256, 0, stream>>>(x2, cpe2_w, cpe2_b, x2, x3, BN, CC, HH, WWD, HH, WWD, 3, 1, 1);
    // 14) t2 = LN(x3) -> bf16 token-major
    layernorm_pack_kernel<<<NTOK / 8, 256, 0, stream>>>(x3, norm2_g, norm2_b, tok, SPX);
    // 15) h = gelu(t2 @ fc1^T + b) -> bf16 [M,1024]
    gemm_wmma_kernel<<<dim3(1024 / 64, NTOK / 128), 256, 0, stream>>>(
        tok, wf1, fc1_b, NTOK, 1024, 256, 256, 1024, 3, SPX, nullptr, nullptr, nullptr, hbuf);
    // 16) out = x3 + h @ fc2^T + b  (final, NCHW f32)
    gemm_wmma_kernel<<<dim3(256 / 64, NTOK / 128), 256, 0, stream>>>(
        hbuf, wf2, fc2_b, NTOK, 256, 1024, 1024, 256, 2, SPX, out, nullptr, x3, nullptr);
}
